// WeightedMultiHeadGAT_10264971837706
// MI455X (gfx1250) — matrix operand order repaired and checked by reading, not compiled, not run on hardware
//
#include <hip/hip_runtime.h>
#include <math.h>

#define NNODE 2048
#define NIN   256
#define NH    12
#define ND    32
#define NHD   384   // H*D
#define NOUT  384

typedef _Float16 half_t;
typedef half_t v16h __attribute__((ext_vector_type(16)));
typedef float  v8f  __attribute__((ext_vector_type(8)));

// ---------------------------------------------------------------------------
// WMMA helpers: f32 = f16 A(16x32) * f16 B(32x16) + f32 C(16x16), wave32.
// A-layout (ISA 7.12.2): lane l<16 owns row l, K={0..7,16..23};
//                        lane l>=16 owns row l-16, K={8..15,24..31}.
// B-layout: lane l<16 owns col l, K={0..7,16..23}; lane l>=16 owns col l-16, K={8..15,24..31}.
// C/D-layout: lane l<16 rows 0..7 (c[r]=row r), lane>=16 rows 8..15; col=l&15.
// ---------------------------------------------------------------------------
__device__ __forceinline__ v8f wmma16(v16h a, v16h b, v8f c) {
  return __builtin_amdgcn_wmma_f32_16x16x32_f16(false, a, false, b,
                                                (short)0, c, false, false);
}

// A operand from LDS row-major tile [16][stride] (stride in halfs, 36 => 8B aligned runs)
__device__ __forceinline__ v16h lds_load_A(const half_t* base, int stride) {
  int lane = threadIdx.x & 31;
  int l = lane & 15, hi = lane >> 4;
  const half_t* p = base + l * stride + hi * 8;
  v16h a;
#pragma unroll
  for (int i = 0; i < 8; ++i) a[i] = p[i];
#pragma unroll
  for (int i = 0; i < 8; ++i) a[8 + i] = p[16 + i];
  return a;
}

// B operand from LDS column-major tile [16 cols][stride K] (stride in halfs)
__device__ __forceinline__ v16h lds_load_B(const half_t* base, int stride) {
  int lane = threadIdx.x & 31;
  int l = lane & 15, hi = lane >> 4;
  const half_t* p = base + l * stride + hi * 8;
  v16h b;
#pragma unroll
  for (int i = 0; i < 8; ++i) b[i] = p[i];
#pragma unroll
  for (int i = 0; i < 8; ++i) b[8 + i] = p[16 + i];
  return b;
}

// ---------------------------------------------------------------------------
// gfx1250 async global->LDS copies (ASYNCcnt-tracked, no VGPR round trip).
// Generic shared-pointer low 32 bits are the wave-relative LDS byte offset
// (ISA 10.2).
// ---------------------------------------------------------------------------
__device__ __forceinline__ void async_ld_b128(void* lds, const void* gptr) {
  unsigned lo = (unsigned)(unsigned long long)lds;
  unsigned long long ga = (unsigned long long)gptr;
  asm volatile("global_load_async_to_lds_b128 %0, %1, off"
               :: "v"(lo), "v"(ga) : "memory");
}
__device__ __forceinline__ void async_ld_b32(void* lds, const void* gptr) {
  unsigned lo = (unsigned)(unsigned long long)lds;
  unsigned long long ga = (unsigned long long)gptr;
  asm volatile("global_load_async_to_lds_b32 %0, %1, off"
               :: "v"(lo), "v"(ga) : "memory");
}
__device__ __forceinline__ void wait_async0() {
  asm volatile("s_wait_asynccnt 0x0" ::: "memory");
}

// ---------------------------------------------------------------------------
// Kernel 1: h = x @ W.T ; res = x @ res_W.T + res_b
// grid (6 colblocks, 128 rowblocks), block 256 (8 waves); wave tile 16x16, K=256.
// ---------------------------------------------------------------------------
__global__ void gat_proj(const float* __restrict__ x, const float* __restrict__ W,
                         const float* __restrict__ resW, const float* __restrict__ res_b,
                         float* __restrict__ h, float* __restrict__ res) {
  __shared__ half_t aT[16 * 36];
  __shared__ half_t bT[8][16 * 36];

  const int row0 = blockIdx.y * 16;
  const int w    = threadIdx.x >> 5;
  const int col0 = blockIdx.x * 128 + w * 16;
  const bool isRes = (col0 >= NHD);
  const float* Wm = isRes ? resW : W;
  const int wcol0 = isRes ? (col0 - NHD) : col0;
  const int lane = threadIdx.x & 31;
  const int l = lane & 15, hi = lane >> 4;

  v8f acc = {};
  for (int k0 = 0; k0 < NIN; k0 += 32) {
    __syncthreads();
    {  // stage A tile (shared by all 8 waves): 16 rows x 32 k, f32->f16
      int e = threadIdx.x * 2;
      int r = e >> 5, k = e & 31;
      const float2 v = *(const float2*)(x + (row0 + r) * NIN + k0 + k);
      aT[r * 36 + k]     = (half_t)v.x;
      aT[r * 36 + k + 1] = (half_t)v.y;
    }
    {  // stage B tile (per wave), column-major: bT[col][k] = Wm[wcol0+col][k0+k]
      const float4* src4 = (const float4*)(Wm + (wcol0 + l) * NIN + k0 + hi * 16);
      half_t* dst = &bT[w][l * 36 + hi * 16];
#pragma unroll
      for (int q = 0; q < 4; ++q) {
        float4 v = src4[q];
        dst[q * 4 + 0] = (half_t)v.x;
        dst[q * 4 + 1] = (half_t)v.y;
        dst[q * 4 + 2] = (half_t)v.z;
        dst[q * 4 + 3] = (half_t)v.w;
      }
    }
    __syncthreads();
    acc = wmma16(lds_load_A(aT, 36), lds_load_B(bT[w], 36), acc);
  }

  // branchless epilogue: bias + destination select hoisted out of the loop
  const int gc = wcol0 + l;
  const float bias = isRes ? res_b[gc] : 0.0f;
  float* dstBase = (isRes ? res : h) + gc;
#pragma unroll
  for (int r = 0; r < 8; ++r) {
    int grow = row0 + hi * 8 + r;
    dstBase[grow * NHD] = acc[r] + bias;
  }
}

// ---------------------------------------------------------------------------
// Kernel 2: a_src[n,h] = sum_d h[n,h,d]*attn_src[h,d]; same for a_dst.
// ---------------------------------------------------------------------------
__global__ void gat_attnvec(const float* __restrict__ h,
                            const float* __restrict__ attn_src,
                            const float* __restrict__ attn_dst,
                            float* __restrict__ a_src, float* __restrict__ a_dst) {
  int idx = blockIdx.x * blockDim.x + threadIdx.x;
  if (idx >= NNODE * NH) return;
  int n = idx / NH, hh = idx % NH;
  const float* hp = h + n * NHD + hh * ND;
  float ss = 0.f, sd = 0.f;
#pragma unroll
  for (int d = 0; d < ND; ++d) {
    float hv = hp[d];
    ss += hv * attn_src[hh * ND + d];
    sd += hv * attn_dst[hh * ND + d];
  }
  a_src[idx] = ss;
  a_dst[idx] = sd;
}

// ---------------------------------------------------------------------------
// Kernel 3: flash-style GAT aggregation + residual (in place over `res`).
// grid (3 head-groups, 128 i-blocks), block 128 (4 waves, 1 head per wave).
// adj & a_dst tiles staged via gfx1250 async global->LDS; h tile staged through
// VGPRs (needs f32->f16 cvt). Online softmax in WMMA A-register layout;
// 2 WMMAs per head per 32-wide j-tile.
// ---------------------------------------------------------------------------
__global__ void gat_agg(const float* __restrict__ h, const float* __restrict__ adj,
                        const float* __restrict__ a_src, const float* __restrict__ a_dst,
                        const float* __restrict__ edge_W, const float* __restrict__ edge_b,
                        float* __restrict__ hout /* == res buffer, in-out */) {
  __shared__ half_t hT[4][32][36];   // [local head][d (B col)][j (K)]  column-major B
  __shared__ float adjT[16][32];
  __shared__ float adstT[4][32];

  const int t = threadIdx.x;
  const int w = t >> 5, lane = t & 31;
  const int l = lane & 15, hi = lane >> 4;
  const int hg = blockIdx.x;            // head group (4 heads)
  const int i0 = blockIdx.y * 16;       // dest-row block
  const int hh = hg * 4 + w;

  const float ew = edge_W[hh];
  const float eb = edge_b[hh];
  const float asrc_row = a_src[(i0 + l) * NH + hh];  // lanes l and l+16 share row l

  // per-thread async-copy coordinates (constant across j-tiles)
  const int ar = t >> 3;            // adj tile row 0..15
  const int ac = (t & 7) * 4;       // adj tile col {0,4,...,28}, 16B chunks
  const int dh = t & 3;             // a_dst local head
  const int dj = t >> 2;            // a_dst j 0..31

  float m_row = -1e30f, s_row = 0.f;
  v8f acc0 = {}, acc1 = {};

  for (int j0 = 0; j0 < NNODE; j0 += 32) {
    __syncthreads();
    // async raw copies: adj tile (16x32 f32, 16B/lane) + a_dst tile (4B/lane)
    async_ld_b128(&adjT[ar][ac], adj + (size_t)(i0 + ar) * NNODE + j0 + ac);
    async_ld_b32(&adstT[dh][dj], a_dst + (size_t)(j0 + dj) * NH + hg * 4 + dh);
    {  // stage h tile f32->f16 into B layout: cols hg*128 + t
      const float* src = h + (size_t)j0 * NHD + hg * 128 + t;
      half_t* dst = &hT[t >> 5][t & 31][0];
#pragma unroll 4
      for (int j = 0; j < 32; ++j) dst[j] = (half_t)src[j * NHD];
    }
    if (j0 + 32 < NNODE) {  // warm L2 for the next tile while we compute
      __builtin_prefetch(adj + (size_t)(i0 + ar) * NNODE + j0 + 32 + ac, 0, 1);
      __builtin_prefetch(h + (size_t)(j0 + 32) * NHD + hg * 128 + t, 0, 1);
    }
    wait_async0();
    __syncthreads();

    // logits in A-register layout: lane pair (l, l+16) owns row l
    const int jb = hi * 8;
    float logit[16];
    float tmax = -3.0e38f;
#pragma unroll
    for (int i = 0; i < 16; ++i) {
      int jj = (i < 8) ? (jb + i) : (jb + 8 + i);  // {jb..jb+7, jb+16..jb+23}
      float av = adjT[l][jj];
      float L;
      if (av > 0.f) {
        L = asrc_row + adstT[w][jj] + av * ew + eb;
        L = (L >= 0.f) ? L : 0.2f * L;             // leaky relu
      } else {
        L = -1e30f;                                // masked
      }
      logit[i] = L;
      tmax = fmaxf(tmax, L);
    }
    tmax = fmaxf(tmax, __shfl_xor(tmax, 16, 32));  // full-row max
    float m_new = fmaxf(m_row, tmax);
    float alpha = __expf(m_row - m_new);
    m_row = m_new;

    v16h pA;
    float ps = 0.f;
#pragma unroll
    for (int i = 0; i < 16; ++i) {
      float p = __expf(logit[i] - m_new);
      ps += p;
      pA[i] = (half_t)p;
    }
    ps += __shfl_xor(ps, 16, 32);
    s_row = s_row * alpha + ps;

    // rescale accumulator: acc row = hi*8 + r, its alpha lives in lane hi*8+r
#pragma unroll
    for (int r = 0; r < 8; ++r) {
      float ar2 = __shfl(alpha, hi * 8 + r, 32);
      acc0[r] *= ar2;
      acc1[r] *= ar2;
    }

    v16h b0 = lds_load_B(&hT[w][0][0], 36);   // d = 0..15
    v16h b1 = lds_load_B(&hT[w][16][0], 36);  // d = 16..31
    acc0 = wmma16(pA, b0, acc0);
    acc1 = wmma16(pA, b1, acc1);
  }

  // finalize: divide by softmax denom, add residual in place
#pragma unroll
  for (int r = 0; r < 8; ++r) {
    float sr = __shfl(s_row, hi * 8 + r, 32);
    float inv = 1.0f / sr;
    int grow = i0 + hi * 8 + r;
    float* pr = hout + (size_t)grow * NHD + hh * ND + l;
    pr[0]  = acc0[r] * inv + pr[0];
    pr[16] = acc1[r] * inv + pr[16];
  }
}

// ---------------------------------------------------------------------------
// Kernel 4: out = gelu(hout @ fus_W.T + fus_b), exact gelu.
// grid (3 colblocks, 128 rowblocks), block 256 (8 waves); K = 384.
// ---------------------------------------------------------------------------
__global__ void gat_fuse(const float* __restrict__ hout, const float* __restrict__ fusW,
                         const float* __restrict__ fus_b, float* __restrict__ out) {
  __shared__ half_t aT[16 * 36];
  __shared__ half_t bT[8][16 * 36];

  const int row0 = blockIdx.y * 16;
  const int w    = threadIdx.x >> 5;
  const int col0 = blockIdx.x * 128 + w * 16;
  const int lane = threadIdx.x & 31;
  const int l = lane & 15, hi = lane >> 4;

  v8f acc = {};
  for (int k0 = 0; k0 < NHD; k0 += 32) {
    __syncthreads();
    {
      int e = threadIdx.x * 2;
      int r = e >> 5, k = e & 31;
      const float2 v = *(const float2*)(hout + (row0 + r) * NHD + k0 + k);
      aT[r * 36 + k]     = (half_t)v.x;
      aT[r * 36 + k + 1] = (half_t)v.y;
    }
    {
      const float4* src4 = (const float4*)(fusW + (col0 + l) * NHD + k0 + hi * 16);
      half_t* dst = &bT[w][l * 36 + hi * 16];
#pragma unroll
      for (int q = 0; q < 4; ++q) {
        float4 v = src4[q];
        dst[q * 4 + 0] = (half_t)v.x;
        dst[q * 4 + 1] = (half_t)v.y;
        dst[q * 4 + 2] = (half_t)v.z;
        dst[q * 4 + 3] = (half_t)v.w;
      }
    }
    __syncthreads();
    acc = wmma16(lds_load_A(aT, 36), lds_load_B(bT[w], 36), acc);
  }

#pragma unroll
  for (int r = 0; r < 8; ++r) {
    int grow = row0 + hi * 8 + r;
    int gc = col0 + l;
    float v = acc[r] + fus_b[gc];
    float g = 0.5f * v * (1.0f + erff(v * 0.70710678118654752f));  // exact gelu
    out[grow * NOUT + gc] = g;
  }
}

// ---------------------------------------------------------------------------
extern "C" void kernel_launch(void* const* d_in, const int* in_sizes, int n_in,
                              void* d_out, int out_size, void* d_ws, size_t ws_size,
                              hipStream_t stream) {
  (void)in_sizes; (void)n_in; (void)out_size; (void)ws_size;
  const float* x        = (const float*)d_in[0];
  const float* adj      = (const float*)d_in[1];
  const float* W        = (const float*)d_in[2];
  const float* attn_src = (const float*)d_in[3];
  const float* attn_dst = (const float*)d_in[4];
  const float* edge_W   = (const float*)d_in[5];
  const float* edge_b   = (const float*)d_in[6];
  const float* res_W    = (const float*)d_in[7];
  const float* res_b    = (const float*)d_in[8];
  const float* fus_W    = (const float*)d_in[9];
  const float* fus_b    = (const float*)d_in[10];

  float* ws      = (float*)d_ws;
  float* h_ws    = ws;                            // 2048*384
  float* res_ws  = ws + NNODE * NHD;              // 2048*384 (becomes h_out in place)
  float* asrc_ws = ws + 2 * NNODE * NHD;          // 2048*12
  float* adst_ws = asrc_ws + NNODE * NH;          // 2048*12

  // 1) projections (h and residual) via WMMA
  gat_proj<<<dim3(6, 128), 256, 0, stream>>>(x, W, res_W, res_b, h_ws, res_ws);
  // 2) per-node attention scalars
  gat_attnvec<<<(NNODE * NH + 255) / 256, 256, 0, stream>>>(h_ws, attn_src, attn_dst,
                                                            asrc_ws, adst_ws);
  // 3) fused flash-style softmax aggregation + residual (WMMA + async LDS)
  gat_agg<<<dim3(3, 128), 128, 0, stream>>>(h_ws, adj, asrc_ws, adst_ws,
                                            edge_W, edge_b, res_ws);
  // 4) output projection + exact GELU (WMMA)
  gat_fuse<<<dim3(3, 128), 256, 0, stream>>>(res_ws, fus_W, fus_b, (float*)d_out);
}
